// Model_84490596647648
// MI455X (gfx1250) — matrix operand order repaired and checked by reading, not compiled, not run on hardware
//
#include <hip/hip_runtime.h>
#include <math.h>

typedef _Float16 f16;
typedef __attribute__((ext_vector_type(8)))  _Float16 v8h;
typedef __attribute__((ext_vector_type(16))) _Float16 v16h;
typedef __attribute__((ext_vector_type(8)))  float    v8f;

#define BB   16
#define LL   198
#define DD   256
#define EDM  512
#define NN   16
#define RR   16
#define NLAY 4
#define PP   196
#define IMGS 224
#define PSZ  16
#define ML   (BB*LL)   /* 3168 */
#define MP   (BB*PP)   /* 3136 */
#define NTW  4         /* N-tiles per wave in GEMM */

__device__ __forceinline__ float siluf(float x)    { return x / (1.f + __expf(-x)); }
__device__ __forceinline__ float softplusf(float x){ return x > 20.f ? x : log1pf(__expf(x)); }

// ---------------------------------------------------------------------------
// f16 GEMM: C[M,N] (f32) = A[M,K] @ W[N,K]^T, row-major.
// Requirements: M%16==0, K%32==0, N%64==0 (wave handles 4 N-tiles, reusing A).
// Block = 4 waves; wave w covers N-tiles [ (bx*4+w)*4 .. +3 ], 16 M-rows.
// ---------------------------------------------------------------------------
__global__ void k_gemm_f16(const f16* __restrict__ A, const f16* __restrict__ W,
                           float* __restrict__ C, int M, int N, int K, int acc)
{
    int wave = threadIdx.x >> 5;
    int lane = threadIdx.x & 31;
    int hs   = lane >> 4;                 // half-select (lanes 0-15 vs 16-31)
    int ls   = lane & 15;
    int tn0  = (blockIdx.x * 4 + wave) * NTW;   // first N-tile for this wave
    if (tn0 * 16 >= N) return;                  // wave-uniform early out
    int m0 = blockIdx.y * 16;

    const f16* arow = A + (size_t)(m0 + ls) * K;
    const f16* wrow = W + (size_t)(tn0 * 16 + ls) * K + hs * 16;
    size_t wstride = (size_t)16 * K;            // next N-tile's weight rows

    v8f c0 = {}, c1 = {}, c2 = {}, c3 = {};
    for (int kk = 0; kk < K; kk += 32) {
        // A 16x32 fragment: lanes<16 hold K[kk..+15],
        //                   lanes>=16 hold K[kk+16..+31]
        v8h alo = *(const v8h*)(arow + kk + hs * 16);
        v8h ahi = *(const v8h*)(arow + kk + hs * 16 + 8);
        v16h a;
        #pragma unroll
        for (int i = 0; i < 8; ++i) { a[i] = alo[i]; a[i + 8] = ahi[i]; }
        // B 32x16 fragments: lane holds 16 contiguous K of one weight row
        v16h b0 = *(const v16h*)(wrow + kk);
        v16h b1 = *(const v16h*)(wrow + wstride + kk);
        v16h b2 = *(const v16h*)(wrow + 2 * wstride + kk);
        v16h b3 = *(const v16h*)(wrow + 3 * wstride + kk);
        c0 = __builtin_amdgcn_wmma_f32_16x16x32_f16(false, a, false, b0, (short)0, c0, false, false);
        c1 = __builtin_amdgcn_wmma_f32_16x16x32_f16(false, a, false, b1, (short)0, c1, false, false);
        c2 = __builtin_amdgcn_wmma_f32_16x16x32_f16(false, a, false, b2, (short)0, c2, false, false);
        c3 = __builtin_amdgcn_wmma_f32_16x16x32_f16(false, a, false, b3, (short)0, c3, false, false);
    }
    // C/D layout: lane holds column (ntile*16 + ls), rows m0+hs*8 .. +7
    float* cp = C + (size_t)(m0 + hs * 8) * N + tn0 * 16 + ls;
    if (acc) {
        #pragma unroll
        for (int r = 0; r < 8; ++r) {
            cp[(size_t)r * N]      += c0[r];
            cp[(size_t)r * N + 16] += c1[r];
            cp[(size_t)r * N + 32] += c2[r];
            cp[(size_t)r * N + 48] += c3[r];
        }
    } else {
        #pragma unroll
        for (int r = 0; r < 8; ++r) {
            cp[(size_t)r * N]      = c0[r];
            cp[(size_t)r * N + 16] = c1[r];
            cp[(size_t)r * N + 32] = c2[r];
            cp[(size_t)r * N + 48] = c3[r];
        }
    }
}

// --------------------------- small helper kernels ---------------------------
__global__ void k_cvt(const float* __restrict__ s, f16* __restrict__ d, int n)
{
    int i = blockIdx.x * blockDim.x + threadIdx.x;
    if (i < n) d[i] = (f16)s[i];
}

// dt_w [NL*ED, 16] -> padded f16 [NL*ED, 32] (upper 16 = 0) so K%32 holds
__global__ void k_cvt_dtw(const float* __restrict__ s, f16* __restrict__ d)
{
    int i = blockIdx.x * blockDim.x + threadIdx.x;
    if (i >= NLAY * EDM * 32) return;
    int row = i >> 5, k = i & 31;
    d[i] = (k < RR) ? (f16)s[row * RR + k] : (f16)0.f;
}

// x_proj [NL,48,512] -> padded f16 [NL,64,512] (rows 48..63 = 0) so N%64 holds
__global__ void k_cvt_xproj(const float* __restrict__ s, f16* __restrict__ d)
{
    int i = blockIdx.x * blockDim.x + threadIdx.x;
    if (i >= NLAY * 64 * 512) return;
    int k = i & 511;
    int row = (i >> 9) & 63;
    int lay = i >> 15;
    d[i] = (row < 48) ? (f16)s[((size_t)lay * 48 + row) * 512 + k] : (f16)0.f;
}

// im2col: xp[b*196+p, c*256+ph*16+pw] = x[b,c, (p/14)*16+ph, (p%14)*16+pw]
__global__ void k_im2col(const float* __restrict__ x, f16* __restrict__ xp)
{
    int idx = blockIdx.x * blockDim.x + threadIdx.x;
    if (idx >= MP * 768) return;
    int row = idx / 768, f = idx % 768;
    int b = row / PP, p = row % PP;
    int c = f >> 8, r = f & 255;
    int ph = r >> 4, pw = r & 15;
    int i = p / 14, j = p % 14;
    size_t src = ((size_t)(b * 3 + c) * IMGS + i * PSZ + ph) * IMGS + j * PSZ + pw;
    xp[idx] = (f16)x[src];
}

// h[b,l,:] = concat(cls_head, emb, cls_tail) + PE
__global__ void k_assemble(const float* __restrict__ emb, const float* __restrict__ head,
                           const float* __restrict__ tail, const float* __restrict__ PE,
                           float* __restrict__ h)
{
    int idx = blockIdx.x * blockDim.x + threadIdx.x;
    if (idx >= ML * DD) return;
    int m = idx / DD, d = idx % DD;
    int b = m / LL, l = m % LL;
    float v;
    if (l == 0)            v = head[d];
    else if (l == LL - 1)  v = tail[d];
    else                   v = emb[(size_t)(b * PP + l - 1) * DD + d];
    h[idx] = v + PE[l * DD + d];
}

// row-wise RMSNorm over D=256, output f16 (GEMM operand)
__global__ void k_rmsnorm_f16(const float* __restrict__ x, const float* __restrict__ w,
                              f16* __restrict__ out)
{
    __shared__ float red[DD];
    int row = blockIdx.x, t = threadIdx.x;
    float v = x[(size_t)row * DD + t];
    red[t] = v * v;
    __syncthreads();
    for (int s = 128; s > 0; s >>= 1) { if (t < s) red[t] += red[t + s]; __syncthreads(); }
    float rs = rsqrtf(red[0] / DD + 1e-5f);
    out[(size_t)row * DD + t] = (f16)(v * rs * w[t]);
}

// causal depthwise conv(K=4) + bias + SiLU; writes f32 (scan) and f16 (GEMM)
__global__ void k_conv(const float* __restrict__ xz, const float* __restrict__ cw,
                       const float* __restrict__ cb, float* __restrict__ xc_f,
                       f16* __restrict__ xc_h)
{
    int idx = blockIdx.x * blockDim.x + threadIdx.x;
    if (idx >= ML * EDM) return;
    int m = idx / EDM, e = idx % EDM;
    int b = m / LL, l = m % LL;
    float acc = cb[e];
    #pragma unroll
    for (int k = 0; k < 4; ++k) {
        int ll = l - 3 + k;
        if (ll >= 0) acc += cw[e * 4 + k] * xz[(size_t)(b * LL + ll) * (2 * EDM) + e];
    }
    float s = siluf(acc);
    xc_f[idx] = s;
    xc_h[idx] = (f16)s;
}

// split dbc[.,64] (48 used) into rms-normed delta (f16, K padded to 32), Bm, Cm
__global__ void k_dbc_post(const float* __restrict__ dbc, const float* __restrict__ ln_dt,
                           const float* __restrict__ ln_B, const float* __restrict__ ln_C,
                           f16* __restrict__ delta_h, float* __restrict__ Bm,
                           float* __restrict__ Cm)
{
    int row = blockIdx.x * blockDim.x + threadIdx.x;
    if (row >= ML) return;
    const float* p = dbc + (size_t)row * 64;
    float s0 = 0, s1 = 0, s2 = 0;
    for (int j = 0; j < 16; ++j) {
        s0 += p[j] * p[j]; s1 += p[16 + j] * p[16 + j]; s2 += p[32 + j] * p[32 + j];
    }
    float r0 = rsqrtf(s0 / 16 + 1e-5f);
    float r1 = rsqrtf(s1 / 16 + 1e-5f);
    float r2 = rsqrtf(s2 / 16 + 1e-5f);
    for (int j = 0; j < 16; ++j) {
        delta_h[(size_t)row * 32 + j]      = (f16)(p[j] * r0 * ln_dt[j]);
        delta_h[(size_t)row * 32 + 16 + j] = (f16)0.f;
        Bm[(size_t)row * 16 + j] = p[16 + j] * r1 * ln_B[j];
        Cm[(size_t)row * 16 + j] = p[32 + j] * r2 * ln_C[j];
    }
}

// selective scan: 1 thread per (b,e), 16 states in registers, B/C via LDS.
// grid = 32 blocks (2 per batch), 256 threads
__global__ void k_scan(const float* __restrict__ dt_f, const float* __restrict__ dt_b,
                       const float* __restrict__ A_log, const float* __restrict__ xc_f,
                       const float* __restrict__ Bm, const float* __restrict__ Cm,
                       const float* __restrict__ Dp, const float* __restrict__ xz_f,
                       f16* __restrict__ y_h)
{
    __shared__ float sB[NN], sC[NN];
    int b = blockIdx.x >> 1;
    int e = ((blockIdx.x & 1) << 8) + threadIdx.x;
    float Ae[NN], hst[NN];
    #pragma unroll
    for (int n = 0; n < NN; ++n) { Ae[n] = -__expf(A_log[e * NN + n]); hst[n] = 0.f; }
    float dpe = Dp[e], dbe = dt_b[e];
    for (int l = 0; l < LL; ++l) {
        int m = b * LL + l;
        __syncthreads();
        if (threadIdx.x < NN)             sB[threadIdx.x]      = Bm[m * NN + threadIdx.x];
        else if (threadIdx.x < 2 * NN)    sC[threadIdx.x - NN] = Cm[m * NN + threadIdx.x - NN];
        __syncthreads();
        float dt  = softplusf(dt_f[(size_t)m * EDM + e] + dbe);
        float xcv = xc_f[(size_t)m * EDM + e];
        float zv  = xz_f[(size_t)m * (2 * EDM) + EDM + e];
        float dx  = dt * xcv;
        float y = 0.f;
        #pragma unroll
        for (int n = 0; n < NN; ++n) {
            float hn = __expf(dt * Ae[n]) * hst[n] + dx * sB[n];
            hst[n] = hn;
            y += hn * sC[n];
        }
        y = (y + dpe * xcv) * siluf(zv);
        y_h[(size_t)m * EDM + e] = (f16)y;
    }
}

// final norm + feat + classifier + softmax. 16 blocks x 256 threads.
__global__ void k_head(const float* __restrict__ h, const float* __restrict__ norm_w,
                       const float* __restrict__ cls_w, float* __restrict__ out)
{
    __shared__ float red[DD];
    __shared__ float feat[DD];
    __shared__ float logits[7];
    int b = blockIdx.x, t = threadIdx.x;
    float v0 = h[(size_t)(b * LL) * DD + t];           // h[b, 0, :]
    float v1 = h[(size_t)(LL - 1) * DD + t];           // h[0, L-1, :]
    red[t] = v0 * v0;
    __syncthreads();
    for (int s = 128; s > 0; s >>= 1) { if (t < s) red[t] += red[t + s]; __syncthreads(); }
    float rs0 = rsqrtf(red[0] / DD + 1e-5f);
    __syncthreads();
    red[t] = v1 * v1;
    __syncthreads();
    for (int s = 128; s > 0; s >>= 1) { if (t < s) red[t] += red[t + s]; __syncthreads(); }
    float rs1 = rsqrtf(red[0] / DD + 1e-5f);
    __syncthreads();
    feat[t] = (v0 * rs0 + v1 * rs1) * norm_w[t];
    __syncthreads();
    for (int c = 0; c < 7; ++c) {
        red[t] = feat[t] * cls_w[c * DD + t];
        __syncthreads();
        for (int s = 128; s > 0; s >>= 1) { if (t < s) red[t] += red[t + s]; __syncthreads(); }
        if (t == 0) logits[c] = red[0];
        __syncthreads();
    }
    if (t == 0) {
        float mx = logits[0];
        for (int c = 1; c < 7; ++c) mx = fmaxf(mx, logits[c]);
        float s = 0.f, e[7];
        for (int c = 0; c < 7; ++c) { e[c] = __expf(logits[c] - mx); s += e[c]; }
        for (int c = 0; c < 7; ++c) out[b * 7 + c] = e[c] / s;
    }
}

// ---------------------------------------------------------------------------
extern "C" void kernel_launch(void* const* d_in, const int* in_sizes, int n_in,
                              void* d_out, int out_size, void* d_ws, size_t ws_size,
                              hipStream_t stream)
{
    (void)in_sizes; (void)n_in; (void)out_size; (void)ws_size;
    const float* x        = (const float*)d_in[0];
    const float* emb_w    = (const float*)d_in[1];
    const float* cls_head = (const float*)d_in[2];
    const float* cls_tail = (const float*)d_in[3];
    const float* PE       = (const float*)d_in[4];
    const float* norm_w   = (const float*)d_in[5];
    const float* cls_w    = (const float*)d_in[6];
    const float* in_proj  = (const float*)d_in[7];
    const float* conv_w   = (const float*)d_in[8];
    const float* conv_b   = (const float*)d_in[9];
    const float* x_proj   = (const float*)d_in[10];
    const float* dt_w     = (const float*)d_in[11];
    const float* dt_b     = (const float*)d_in[12];
    const float* A_log    = (const float*)d_in[13];
    const float* Dp       = (const float*)d_in[14];
    const float* out_proj = (const float*)d_in[15];
    const float* ln_in    = (const float*)d_in[16];
    const float* ln_dt    = (const float*)d_in[17];
    const float* ln_B     = (const float*)d_in[18];
    const float* ln_C     = (const float*)d_in[19];
    float* out = (float*)d_out;

    char* wsb = (char*)d_ws;
    size_t off = 0;
    auto alloc = [&](size_t bytes) -> char* {
        char* p = wsb + off;
        off = (off + bytes + 255) & ~size_t(255);
        return p;
    };
    f16*   emb_w_h    = (f16*)alloc((size_t)256 * 768 * 2);
    f16*   in_proj_h  = (f16*)alloc((size_t)NLAY * 1024 * 256 * 2);
    f16*   x_proj_h   = (f16*)alloc((size_t)NLAY * 64 * 512 * 2);   // padded N
    f16*   dtw_h      = (f16*)alloc((size_t)NLAY * 512 * 32 * 2);   // padded K
    f16*   out_proj_h = (f16*)alloc((size_t)NLAY * 256 * 512 * 2);
    f16*   xp_h       = (f16*)alloc((size_t)MP * 768 * 2);
    float* emb_f      = (float*)alloc((size_t)MP * 256 * 4);
    float* h_f        = (float*)alloc((size_t)ML * 256 * 4);
    f16*   xn_h       = (f16*)alloc((size_t)ML * 256 * 2);
    float* xz_f       = (float*)alloc((size_t)ML * 1024 * 4);
    float* xc_f       = (float*)alloc((size_t)ML * 512 * 4);
    f16*   xc_h       = (f16*)alloc((size_t)ML * 512 * 2);
    float* dbc_f      = (float*)alloc((size_t)ML * 64 * 4);         // padded N
    f16*   delta_h    = (f16*)alloc((size_t)ML * 32 * 2);
    float* Bm_f       = (float*)alloc((size_t)ML * 16 * 4);
    float* Cm_f       = (float*)alloc((size_t)ML * 16 * 4);
    float* dt_f       = (float*)alloc((size_t)ML * 512 * 4);
    f16*   y_h        = (f16*)alloc((size_t)ML * 512 * 2);

    // weight conversion (deterministic, runs every call)
    k_cvt<<<(256 * 768 + 255) / 256, 256, 0, stream>>>(emb_w, emb_w_h, 256 * 768);
    k_cvt<<<(NLAY * 1024 * 256 + 255) / 256, 256, 0, stream>>>(in_proj, in_proj_h, NLAY * 1024 * 256);
    k_cvt_xproj<<<(NLAY * 64 * 512 + 255) / 256, 256, 0, stream>>>(x_proj, x_proj_h);
    k_cvt<<<(NLAY * 256 * 512 + 255) / 256, 256, 0, stream>>>(out_proj, out_proj_h, NLAY * 256 * 512);
    k_cvt_dtw<<<(NLAY * 512 * 32 + 255) / 256, 256, 0, stream>>>(dt_w, dtw_h);

    // patch embedding: M=3136, N=256, K=768 -> grid.x = 256/(64*4) = 1
    k_im2col<<<(MP * 768 + 255) / 256, 256, 0, stream>>>(x, xp_h);
    k_gemm_f16<<<dim3(1, MP / 16), 128, 0, stream>>>(xp_h, emb_w_h, emb_f, MP, 256, 768, 0);
    k_assemble<<<(ML * DD + 255) / 256, 256, 0, stream>>>(emb_f, cls_head, cls_tail, PE, h_f);

    for (int i = 0; i < NLAY; ++i) {
        k_rmsnorm_f16<<<ML, DD, 0, stream>>>(h_f, ln_in + i * DD, xn_h);
        // N=1024 -> grid.x = 1024/256 = 4
        k_gemm_f16<<<dim3(4, ML / 16), 128, 0, stream>>>(
            xn_h, in_proj_h + (size_t)i * 1024 * 256, xz_f, ML, 1024, 256, 0);
        k_conv<<<(ML * EDM + 255) / 256, 256, 0, stream>>>(
            xz_f, conv_w + i * EDM * 4, conv_b + i * EDM, xc_f, xc_h);
        // N=64 -> grid.x = 1 (wave 0 active, waves 1-3 early-out)
        k_gemm_f16<<<dim3(1, ML / 16), 128, 0, stream>>>(
            xc_h, x_proj_h + (size_t)i * 64 * 512, dbc_f, ML, 64, 512, 0);
        k_dbc_post<<<(ML + 255) / 256, 256, 0, stream>>>(
            dbc_f, ln_dt + i * RR, ln_B + i * NN, ln_C + i * NN, delta_h, Bm_f, Cm_f);
        // N=512 -> grid.x = 2
        k_gemm_f16<<<dim3(2, ML / 16), 128, 0, stream>>>(
            delta_h, dtw_h + (size_t)i * 512 * 32, dt_f, ML, 512, 32, 0);
        k_scan<<<BB * 2, 256, 0, stream>>>(
            dt_f, dt_b + i * EDM, A_log + (size_t)i * EDM * NN, xc_f, Bm_f, Cm_f,
            Dp + i * EDM, xz_f, y_h);
        // N=256 -> grid.x = 1, accumulate into residual
        k_gemm_f16<<<dim3(1, ML / 16), 128, 0, stream>>>(
            y_h, out_proj_h + (size_t)i * 256 * 512, h_f, ML, 256, 512, 1);
    }
    k_head<<<BB, DD, 0, stream>>>(h_f, norm_w, cls_w, out);
}
